// KDEMixture_88003879895536
// MI455X (gfx1250) — compile-verified, matrix operand order checked by reading, not run on hardware
//
#include <hip/hip_runtime.h>

typedef __attribute__((ext_vector_type(4)))  float          v4f;
typedef __attribute__((ext_vector_type(8)))  float          v8f;
typedef __attribute__((ext_vector_type(8)))  unsigned short u16x8;
typedef __attribute__((ext_vector_type(16))) unsigned short u16x16;
typedef __attribute__((ext_vector_type(16))) __bf16         v16bf;

#define M_TOT 16384
#define N_TOT 16384
#define D_DIM 64
#define NCHUNKS 4

static __device__ __forceinline__ unsigned short f32_to_bf16_rne(float f) {
    unsigned int u = __float_as_uint(f);
    unsigned int r = u + 0x7FFFu + ((u >> 16) & 1u);   // round-to-nearest-even
    return (unsigned short)(r >> 16);
}
static __device__ __forceinline__ float bf16_to_f32(unsigned short h) {
    return __uint_as_float(((unsigned int)h) << 16);
}

// ---------------------------------------------------------------------------
// Precompute: row norms, plus split-bf16 planes.
//   points : decompose (2*p) = P2h + P2l   (exact 2x scale, RNE splits)
//   spatial: decompose  s    = Sh  + Sl
// ---------------------------------------------------------------------------
__global__ void kde_prep_kernel(const float* __restrict__ points,
                                const float* __restrict__ spatial,
                                float* __restrict__ psq, float* __restrict__ ssq,
                                unsigned short* __restrict__ P2h,
                                unsigned short* __restrict__ P2l,
                                unsigned short* __restrict__ Sh,
                                unsigned short* __restrict__ Sl) {
    int i = blockIdx.x * blockDim.x + threadIdx.x;
    if (i < N_TOT) {
        const float* r = spatial + (size_t)i * D_DIM;
        unsigned short* oh = Sh + (size_t)i * D_DIM;
        unsigned short* ol = Sl + (size_t)i * D_DIM;
        float acc = 0.0f;
        #pragma unroll 8
        for (int k = 0; k < D_DIM; ++k) {
            float x = r[k];
            acc += x * x;
            unsigned short h = f32_to_bf16_rne(x);
            oh[k] = h;
            ol[k] = f32_to_bf16_rne(x - bf16_to_f32(h));
        }
        ssq[i] = acc;
    }
    if (i < M_TOT) {
        const float* r = points + (size_t)i * D_DIM;
        unsigned short* oh = P2h + (size_t)i * D_DIM;
        unsigned short* ol = P2l + (size_t)i * D_DIM;
        float acc = 0.0f;
        #pragma unroll 8
        for (int k = 0; k < D_DIM; ++k) {
            float x = r[k];
            acc += x * x;
            float y = 2.0f * x;                         // exact
            unsigned short h = f32_to_bf16_rne(y);
            oh[k] = h;
            ol[k] = f32_to_bf16_rne(y - bf16_to_f32(h));
        }
        psq[i] = acc;
    }
}

// Build a v16bf WMMA operand fragment from a bf16 row:
// per-lane elements  chunk*32 + base + {0..7}  and  chunk*32 + base + 16 + {0..7}
static __device__ __forceinline__ v16bf load_frag(const unsigned short* row,
                                                  int chunk, int base) {
    u16x8 a = *(const u16x8*)(row + chunk * 32 + base);
    u16x8 b = *(const u16x8*)(row + chunk * 32 + base + 16);
    u16x16 c = __builtin_shufflevector(a, b, 0, 1, 2, 3, 4, 5, 6, 7,
                                             8, 9, 10, 11, 12, 13, 14, 15);
    return __builtin_bit_cast(v16bf, c);
}

// ---------------------------------------------------------------------------
// Fused split-bf16 WMMA GEMM + RBF exp + weighted row-sum.
// Each wave32 owns one 16-row m-tile and one N-chunk.
// cross2 = (2p).s  via  pH.sH + pH.sL + pL.sH  (6x V_WMMA_F32_16X16X32_BF16)
// ---------------------------------------------------------------------------
__global__ __launch_bounds__(256) void kde_main_kernel(
    const unsigned short* __restrict__ P2h, const unsigned short* __restrict__ P2l,
    const unsigned short* __restrict__ Sh,  const unsigned short* __restrict__ Sl,
    const float* __restrict__ W, const float* __restrict__ bw,
    const float* __restrict__ psq, const float* __restrict__ ssq,
    float* __restrict__ partial) {
    const int lane = threadIdx.x & 31;
    const int wave = threadIdx.x >> 5;
    const int mblocks = M_TOT / 128;              // 128
    const int mblock = blockIdx.x % mblocks;
    const int chunk  = blockIdx.x / mblocks;      // 0..NCHUNKS-1
    const int mtile  = mblock * 8 + wave;         // 0..1023
    const int half   = lane >> 4;                 // 0 or 1
    const int base   = half << 3;                 // bf16 K sub-offset {0,8}
    const int row    = (mtile << 4) + (lane & 15);

    const float b = bw[0];
    const float nls = -1.44269504088896340736f / (2.0f * b * b); // -log2(e)/(2b^2)

    // Preload A fragments (points tile, pre-scaled by 2 in prep).
    v16bf aH[2], aL[2];
    {
        const unsigned short* ph = P2h + (size_t)row * D_DIM;
        const unsigned short* pl = P2l + (size_t)row * D_DIM;
        #pragma unroll
        for (int c = 0; c < 2; ++c) {
            aH[c] = load_frag(ph, c, base);
            aL[c] = load_frag(pl, c, base);
        }
    }

    // ||p||^2 for the 8 m's this lane owns in the C layout.
    float psq8[8];
    #pragma unroll
    for (int v = 0; v < 8; ++v)
        psq8[v] = psq[(mtile << 4) + (half << 3) + v];

    float acc[8];
    #pragma unroll
    for (int v = 0; v < 8; ++v) acc[v] = 0.0f;

    const int nPerChunk = N_TOT / NCHUNKS;
    const int n0 = chunk * nPerChunk;

    for (int nb = 0; nb < nPerChunk; nb += 16) {
        const int n = n0 + nb + (lane & 15);
        const unsigned short* sh = Sh + (size_t)n * D_DIM;
        const unsigned short* sl = Sl + (size_t)n * D_DIM;
        const float ssn = ssq[n];
        const float wn  = W[n];

        v16bf bH0 = load_frag(sh, 0, base);
        v16bf bH1 = load_frag(sh, 1, base);
        v16bf bL0 = load_frag(sl, 0, base);
        v16bf bL1 = load_frag(sl, 1, base);

        v8f c = {};
        c = __builtin_amdgcn_wmma_f32_16x16x32_bf16(false, aH[0], false, bH0,
                                                    (short)0, c, false, false);
        c = __builtin_amdgcn_wmma_f32_16x16x32_bf16(false, aH[1], false, bH1,
                                                    (short)0, c, false, false);
        c = __builtin_amdgcn_wmma_f32_16x16x32_bf16(false, aH[0], false, bL0,
                                                    (short)0, c, false, false);
        c = __builtin_amdgcn_wmma_f32_16x16x32_bf16(false, aH[1], false, bL1,
                                                    (short)0, c, false, false);
        c = __builtin_amdgcn_wmma_f32_16x16x32_bf16(false, aL[0], false, bH0,
                                                    (short)0, c, false, false);
        c = __builtin_amdgcn_wmma_f32_16x16x32_bf16(false, aL[1], false, bH1,
                                                    (short)0, c, false, false);

        #pragma unroll
        for (int v = 0; v < 8; ++v) {
            float d2 = psq8[v] + ssn - c[v];     // c holds 2*cross (fp32 accum)
            d2 = fmaxf(d2, 0.0f);
            acc[v] += wn * __builtin_amdgcn_exp2f(d2 * nls);
        }
    }

    // Weighted row-sum over n: butterfly within each 16-lane half (wave32).
    #pragma unroll
    for (int v = 0; v < 8; ++v) {
        float s = acc[v];
        s += __shfl_xor(s, 1);
        s += __shfl_xor(s, 2);
        s += __shfl_xor(s, 4);
        s += __shfl_xor(s, 8);
        acc[v] = s;
    }
    if ((lane & 15) == 0) {
        const int mbase = (mtile << 4) + (half << 3);
        #pragma unroll
        for (int v = 0; v < 8; ++v)
            partial[(size_t)chunk * M_TOT + mbase + v] = acc[v];
    }
}

// ---------------------------------------------------------------------------
// Sum the NCHUNKS partial results into d_out.
// ---------------------------------------------------------------------------
__global__ void kde_reduce_kernel(const float* __restrict__ partial,
                                  float* __restrict__ out) {
    int m = blockIdx.x * blockDim.x + threadIdx.x;
    if (m < M_TOT) {
        float s = 0.0f;
        #pragma unroll
        for (int c = 0; c < NCHUNKS; ++c)
            s += partial[(size_t)c * M_TOT + m];
        out[m] = s;
    }
}

extern "C" void kernel_launch(void* const* d_in, const int* in_sizes, int n_in,
                              void* d_out, int out_size, void* d_ws, size_t ws_size,
                              hipStream_t stream) {
    const float* points  = (const float*)d_in[0];   // [M, 64]
    const float* spatial = (const float*)d_in[1];   // [N, 64]
    const float* weights = (const float*)d_in[2];   // [N]
    const float* bandw   = (const float*)d_in[3];   // scalar

    float* psq     = (float*)d_ws;                       // [M]
    float* ssq     = psq + M_TOT;                        // [N]
    float* partial = ssq + N_TOT;                        // [NCHUNKS, M]
    unsigned short* P2h = (unsigned short*)(partial + (size_t)NCHUNKS * M_TOT);
    unsigned short* P2l = P2h + (size_t)M_TOT * D_DIM;
    unsigned short* Sh  = P2l + (size_t)M_TOT * D_DIM;
    unsigned short* Sl  = Sh  + (size_t)N_TOT * D_DIM;

    // 1) row norms + split-bf16 planes
    kde_prep_kernel<<<(N_TOT + 255) / 256, 256, 0, stream>>>(
        points, spatial, psq, ssq, P2h, P2l, Sh, Sl);

    // 2) fused bf16-split WMMA GEMM + exp + weighted reduction over n
    const int grid = (M_TOT / 128) * NCHUNKS;   // 512 blocks x 256 threads = 4096 waves
    kde_main_kernel<<<grid, 256, 0, stream>>>(P2h, P2l, Sh, Sl, weights, bandw,
                                              psq, ssq, partial);

    // 3) chunk reduction
    kde_reduce_kernel<<<(M_TOT + 255) / 256, 256, 0, stream>>>(partial, (float*)d_out);
}
